// BaselineGCN2_33303176413850
// MI455X (gfx1250) — matrix operand order repaired and checked by reading, not compiled, not run on hardware
//
#include <hip/hip_runtime.h>
#include <hip/hip_bf16.h>

#define THREADS 256

typedef __attribute__((ext_vector_type(16))) __bf16 v16bf;
typedef __attribute__((ext_vector_type(8)))  float  v8f;

union Frag { v16bf v; uint4 q[2]; };

__device__ __forceinline__ v8f wmma_bf16(const Frag& a, const Frag& b, v8f c) {
  // 8 args: (neg_a, A, neg_b, B, c_mod, C, reuse_a, reuse_b)
  return __builtin_amdgcn_wmma_f32_16x16x32_bf16(false, a.v, false, b.v, (short)0, c, false, false);
}

// ---------------- utility kernels ----------------
__global__ void k_f32_to_bf16(const float* __restrict__ in, __bf16* __restrict__ out, int n) {
  int i = blockIdx.x * THREADS + threadIdx.x;
  if (i < n) out[i] = (__bf16)in[i];
}

// W: [K, Nn] row-major f32 -> Wt: [Nn, K] row-major bf16 (N-major, matches B-fragment layout)
__global__ void k_transpose_bf16(const float* __restrict__ W, __bf16* __restrict__ Wt, int K, int Nn) {
  int i = blockIdx.x * THREADS + threadIdx.x;
  if (i < K * Nn) {
    int k = i / Nn, n = i % Nn;
    Wt[(size_t)n * K + k] = (__bf16)W[i];
  }
}

__global__ void k_deg_init(float* deg, int N) {
  int i = blockIdx.x * THREADS + threadIdx.x;
  if (i < N) deg[i] = 1.0f;   // self loop
}
__global__ void k_deg_scatter(const int* __restrict__ dst, float* deg, int E) {
  int i = blockIdx.x * THREADS + threadIdx.x;
  if (i < E) atomicAdd(&deg[dst[i]], 1.0f);
}
__global__ void k_deg_rsqrt(float* deg, int N) {
  int i = blockIdx.x * THREADS + threadIdx.x;
  if (i < N) deg[i] = rsqrtf(deg[i]);
}

// acc[i,c] = h[i,c] * dinv[i]^2   (self-loop term, also initializes acc)
__global__ void k_gcn_self(const float* __restrict__ h, const float* __restrict__ dinv,
                           float* __restrict__ acc, int N) {
  int i = blockIdx.x * THREADS + threadIdx.x;
  if (i < N * 256) {
    float di = dinv[i >> 8];
    acc[i] = h[i] * di * di;
  }
}

// acc[dst] += h[src] * dinv[src]*dinv[dst]  — one thread per (edge, 4 features)
__global__ void k_gcn_scatter(const float* __restrict__ h, const int* __restrict__ src,
                              const int* __restrict__ dst, const float* __restrict__ dinv,
                              float* __restrict__ acc, int E) {
  int t = blockIdx.x * THREADS + threadIdx.x;
  if (t >= E * 64) return;
  int e = t >> 6;
  int c = (t & 63) << 2;
  int s = src[e], d = dst[e];
  float nrm = dinv[s] * dinv[d];
  const float4 v = *(const float4*)(h + (size_t)s * 256 + c);
  float* p = acc + (size_t)d * 256 + c;
  atomicAdd(p + 0, v.x * nrm);
  atomicAdd(p + 1, v.y * nrm);
  atomicAdd(p + 2, v.z * nrm);
  atomicAdd(p + 3, v.w * nrm);
}

// act = relu(acc + b) -> bf16 (and optionally float copy for pooling)
__global__ void k_gcn_post(const float* __restrict__ acc, const float* __restrict__ bias,
                           __bf16* __restrict__ actb, float* __restrict__ xg, int N) {
  int i = blockIdx.x * THREADS + threadIdx.x;
  if (i < N * 256) {
    float v = fmaxf(acc[i] + bias[i & 255], 0.0f);
    actb[i] = (__bf16)v;
    if (xg) xg[i] = v;
  }
}

__global__ void k_pool(const float* __restrict__ xg, const float* __restrict__ xe,
                       const int* __restrict__ batch, float* __restrict__ g, int N) {
  int i = blockIdx.x * THREADS + threadIdx.x;
  if (i >= N * 512) return;
  int node = i >> 9, c = i & 511;
  float v = (c < 256) ? xg[(size_t)node * 256 + c] : xe[(size_t)node * 256 + (c - 256)];
  atomicAdd(&g[(size_t)batch[node] * 512 + c], v);
}

__global__ void k_head_out(const float* __restrict__ h1, const float* __restrict__ ow,
                           const float* __restrict__ ob, float* __restrict__ out, int G) {
  int i = blockIdx.x * blockDim.x + threadIdx.x;
  if (i < G) {
    float s = ob[0];
    for (int k = 0; k < 256; ++k) s += h1[(size_t)i * 256 + k] * ow[k];
    out[i] = s;
  }
}

// ---------------- generic WMMA GEMM: C[M,Nn] = A[M,K] @ B (Bt = [Nn,K] bf16) ----------------
// block = 256 threads (8 waves), tile 128x64; grid = (ceil(M/128), Nn/64)
__global__ void k_gemm(const __bf16* __restrict__ A, const __bf16* __restrict__ Bt,
                       const float* __restrict__ bias, float* __restrict__ Cf,
                       __bf16* __restrict__ Cb, int M, int K, int Nn, int relu) {
  const int lane = threadIdx.x & 31;
  const int wave = threadIdx.x >> 5;
  const int m0 = blockIdx.x * 128 + wave * 16;
  const int n0 = blockIdx.y * 64;
  const int lm = lane & 15, lh = lane >> 4;
  int arow = m0 + lm; if (arow >= M) arow = M - 1;       // clamp loads; EXEC stays full for WMMA
  const __bf16* arp = A + (size_t)arow * K;
  v8f acc0 = {}, acc1 = {}, acc2 = {}, acc3 = {};
  for (int kk = 0; kk < K; kk += 32) {
    Frag a;
    a.q[0] = *(const uint4*)(arp + kk + lh * 8);         // K = kk+{0..7 | 8..15}
    a.q[1] = *(const uint4*)(arp + kk + 16 + lh * 8);    // K = kk+16+{0..7 | 8..15}
    const __bf16* bp = Bt + (size_t)(n0 + lm) * K + kk + lh * 8;
#pragma unroll
    for (int j = 0; j < 4; ++j) {
      Frag b;
      b.q[0] = *(const uint4*)(bp + (size_t)j * 16 * K);
      b.q[1] = *(const uint4*)(bp + (size_t)j * 16 * K + 16);
      v8f& c = (j == 0 ? acc0 : j == 1 ? acc1 : j == 2 ? acc2 : acc3);
      c = wmma_bf16(a, b, c);
    }
  }
#pragma unroll
  for (int j = 0; j < 4; ++j) {
    v8f c = (j == 0 ? acc0 : j == 1 ? acc1 : j == 2 ? acc2 : acc3);
    int col = n0 + j * 16 + lm;
    float bv = bias ? bias[col] : 0.0f;
#pragma unroll
    for (int v = 0; v < 8; ++v) {
      int row = m0 + v + lh * 8;
      if (row < M) {
        float val = c[v] + bv;
        if (relu) val = fmaxf(val, 0.0f);
        if (Cf) Cf[(size_t)row * Nn + col] = val;
        if (Cb) Cb[(size_t)row * Nn + col] = (__bf16)val;
      }
    }
  }
}

// ---------------- fused EdgeConv: relu([x_i, x_j-x_i] @ W1 + b1) @ W2 + b2, max-agg ----------------
// D = input feature dim (128 or 256). block = 256 threads, 64 edges / block.
template <int D>
__global__ void k_edgeconv(const __bf16* __restrict__ xin,
                           const int* __restrict__ src, const int* __restrict__ dst,
                           const __bf16* __restrict__ W1t, const float* __restrict__ b1,
                           const __bf16* __restrict__ W2t, const float* __restrict__ b2,
                           float* __restrict__ out, int E) {
  constexpr int K1 = 2 * D;
  __shared__ alignas(16) __bf16 sA[64 * K1];    // gathered edge features
  __shared__ alignas(16) __bf16 sM[64 * 256];   // relu'd hidden
  __shared__ int sDst[64];
  const int tid = threadIdx.x;
  const int lane = tid & 31, wave = tid >> 5;
  const int e0 = blockIdx.x * 64;

  if (tid < 64) sDst[tid] = dst[min(e0 + tid, E - 1)];
  for (int idx = tid; idx < 64 * D; idx += THREADS) {
    int e = idx / D, c = idx % D;
    int eg = min(e0 + e, E - 1);
    int s = src[eg], d = dst[eg];
    float xs = (float)xin[(size_t)s * D + c];
    float xd = (float)xin[(size_t)d * D + c];
    sA[e * K1 + c] = (__bf16)xd;
    sA[e * K1 + D + c] = (__bf16)(xs - xd);
  }
  __syncthreads();

  const int lm = lane & 15, lh = lane >> 4;
  const int mrow = (wave & 3) * 16;       // 4 M-tiles
  const int nbase = (wave >> 2) * 128;    // 2 N-halves, 8 n-tiles each

  // GEMM1: sM = relu(sA @ W1 + b1)
  {
    v8f acc[8] = {};
    for (int kk = 0; kk < K1; kk += 32) {
      Frag a;
      const __bf16* ap = sA + (mrow + lm) * K1 + kk + lh * 8;
      a.q[0] = *(const uint4*)ap;
      a.q[1] = *(const uint4*)(ap + 16);
#pragma unroll
      for (int j = 0; j < 8; ++j) {
        Frag b;
        const __bf16* bp = W1t + (size_t)(nbase + j * 16 + lm) * K1 + kk + lh * 8;
        b.q[0] = *(const uint4*)bp;
        b.q[1] = *(const uint4*)(bp + 16);
        acc[j] = wmma_bf16(a, b, acc[j]);
      }
    }
#pragma unroll
    for (int j = 0; j < 8; ++j) {
      int col = nbase + j * 16 + lm;
      float bv = b1[col];
#pragma unroll
      for (int v = 0; v < 8; ++v) {
        int row = mrow + v + lh * 8;
        sM[row * 256 + col] = (__bf16)fmaxf(acc[j][v] + bv, 0.0f);
      }
    }
  }
  __syncthreads();

  // GEMM2: msg = sM @ W2 + b2 -> atomic max into out[dst]
  {
    v8f acc[8] = {};
    for (int kk = 0; kk < 256; kk += 32) {
      Frag a;
      const __bf16* ap = sM + (mrow + lm) * 256 + kk + lh * 8;
      a.q[0] = *(const uint4*)ap;
      a.q[1] = *(const uint4*)(ap + 16);
#pragma unroll
      for (int j = 0; j < 8; ++j) {
        Frag b;
        const __bf16* bp = W2t + (size_t)(nbase + j * 16 + lm) * 256 + kk + lh * 8;
        b.q[0] = *(const uint4*)bp;
        b.q[1] = *(const uint4*)(bp + 16);
        acc[j] = wmma_bf16(a, b, acc[j]);
      }
    }
#pragma unroll
    for (int j = 0; j < 8; ++j) {
      int col = nbase + j * 16 + lm;
      float bv = b2[col];
#pragma unroll
      for (int v = 0; v < 8; ++v) {
        int row = mrow + v + lh * 8;
        float val = acc[j][v] + bv;
        // final = relu(max over edges), empty -> 0: buffer is 0-init, only positive
        // candidates matter, so uint bit-pattern max is exact for non-negative floats.
        if (val > 0.0f) {
          int node = sDst[row];
          atomicMax((unsigned int*)(out + (size_t)node * 256 + col), __float_as_uint(val));
        }
      }
    }
  }
}

// ---------------- host-side orchestration ----------------
extern "C" void kernel_launch(void* const* d_in, const int* in_sizes, int n_in,
                              void* d_out, int out_size, void* d_ws, size_t ws_size,
                              hipStream_t stream) {
  (void)n_in; (void)ws_size;
  const float* x      = (const float*)d_in[0];
  const int*   eidx   = (const int*)  d_in[1];
  const int*   batch  = (const int*)  d_in[2];
  const float* gcn_w[4] = {(const float*)d_in[3], (const float*)d_in[5],
                           (const float*)d_in[7], (const float*)d_in[9]};
  const float* gcn_b[4] = {(const float*)d_in[4], (const float*)d_in[6],
                           (const float*)d_in[8], (const float*)d_in[10]};
  const float* ec1_w1 = (const float*)d_in[11]; const float* ec1_b1 = (const float*)d_in[12];
  const float* ec1_w2 = (const float*)d_in[13]; const float* ec1_b2 = (const float*)d_in[14];
  const float* ec2_w1 = (const float*)d_in[15]; const float* ec2_b1 = (const float*)d_in[16];
  const float* ec2_w2 = (const float*)d_in[17]; const float* ec2_b2 = (const float*)d_in[18];
  const float* fc1_w  = (const float*)d_in[19]; const float* fc1_b  = (const float*)d_in[20];
  const float* out_w  = (const float*)d_in[21]; const float* out_b  = (const float*)d_in[22];
  float* outp = (float*)d_out;

  const int N  = in_sizes[0] / 128;
  const int E  = in_sizes[1] / 2;
  const int Gn = out_size;                  // 64 graphs
  const int* src  = eidx;
  const int* dstp = eidx + E;

  char* ws = (char*)d_ws;
  size_t off = 0;
  auto alloc = [&](size_t bytes) -> char* {
    size_t o = off; off += (bytes + 255) & ~(size_t)255; return ws + o;
  };
  __bf16* xb    = (__bf16*)alloc((size_t)N * 128 * 2);   // x in bf16 (GCN L1 + EC1 input)
  __bf16* actb  = (__bf16*)alloc((size_t)N * 256 * 2);   // bf16 activation (reused)
  float*  hbuf  = (float*) alloc((size_t)N * 256 * 4);   // GCN gemm out / EC1 out
  float*  accb  = (float*) alloc((size_t)N * 256 * 4);   // GCN aggregation / EC2 out
  float*  dinv  = (float*) alloc((size_t)N * 4);
  float*  xg    = (float*) alloc((size_t)N * 256 * 4);   // GCN branch result
  __bf16* w1t   = (__bf16*)alloc((size_t)128 * 256 * 2);
  __bf16* w2t   = (__bf16*)alloc((size_t)256 * 256 * 2);
  __bf16* w3t   = (__bf16*)alloc((size_t)256 * 256 * 2);
  __bf16* w4t   = (__bf16*)alloc((size_t)256 * 256 * 2);
  __bf16* e1w1t = (__bf16*)alloc((size_t)256 * 256 * 2);
  __bf16* e1w2t = (__bf16*)alloc((size_t)256 * 256 * 2);
  __bf16* e2w1t = (__bf16*)alloc((size_t)512 * 256 * 2);
  __bf16* e2w2t = (__bf16*)alloc((size_t)256 * 256 * 2);
  __bf16* fc1t  = (__bf16*)alloc((size_t)512 * 256 * 2);
  float*  gpool = (float*) alloc((size_t)Gn * 512 * 4);
  __bf16* gb    = (__bf16*)alloc((size_t)Gn * 512 * 2);
  float*  h1f   = (float*) alloc((size_t)Gn * 256 * 4);

  const int nb33 = (128 * 256 + 255) / 256;
  const int nb66 = (256 * 256 + 255) / 256;
  const int nbC6 = (512 * 256 + 255) / 256;
  k_transpose_bf16<<<nb33, THREADS, 0, stream>>>(gcn_w[0], w1t, 128, 256);
  k_transpose_bf16<<<nb66, THREADS, 0, stream>>>(gcn_w[1], w2t, 256, 256);
  k_transpose_bf16<<<nb66, THREADS, 0, stream>>>(gcn_w[2], w3t, 256, 256);
  k_transpose_bf16<<<nb66, THREADS, 0, stream>>>(gcn_w[3], w4t, 256, 256);
  k_transpose_bf16<<<nb66, THREADS, 0, stream>>>(ec1_w1, e1w1t, 256, 256);
  k_transpose_bf16<<<nb66, THREADS, 0, stream>>>(ec1_w2, e1w2t, 256, 256);
  k_transpose_bf16<<<nbC6, THREADS, 0, stream>>>(ec2_w1, e2w1t, 512, 256);
  k_transpose_bf16<<<nb66, THREADS, 0, stream>>>(ec2_w2, e2w2t, 256, 256);
  k_transpose_bf16<<<nbC6, THREADS, 0, stream>>>(fc1_w, fc1t, 512, 256);

  k_f32_to_bf16<<<(N * 128 + 255) / 256, THREADS, 0, stream>>>(x, xb, N * 128);

  k_deg_init   <<<(N + 255) / 256, THREADS, 0, stream>>>(dinv, N);
  k_deg_scatter<<<(E + 255) / 256, THREADS, 0, stream>>>(dstp, dinv, E);
  k_deg_rsqrt  <<<(N + 255) / 256, THREADS, 0, stream>>>(dinv, N);

  const dim3 gemmGrid((N + 127) / 128, 4);
  const int nbNH = (N * 256 + 255) / 256;
  const int nbSc = (int)(((size_t)E * 64 + 255) / 256);

  // ---- GCN branch ----
  const __bf16* gact = xb; int gK = 128;
  const __bf16* wts[4] = {w1t, w2t, w3t, w4t};
  for (int l = 0; l < 4; ++l) {
    k_gemm<<<gemmGrid, THREADS, 0, stream>>>(gact, wts[l], nullptr, hbuf,
                                             (__bf16*)nullptr, N, gK, 256, 0);
    k_gcn_self   <<<nbNH, THREADS, 0, stream>>>(hbuf, dinv, accb, N);
    k_gcn_scatter<<<nbSc, THREADS, 0, stream>>>(hbuf, src, dstp, dinv, accb, E);
    k_gcn_post   <<<nbNH, THREADS, 0, stream>>>(accb, gcn_b[l], actb,
                                                (l == 3) ? xg : nullptr, N);
    gact = actb; gK = 256;
  }

  // ---- EdgeConv branch ----
  hipMemsetAsync(hbuf, 0, (size_t)N * 256 * 4, stream);
  k_edgeconv<128><<<(E + 63) / 64, THREADS, 0, stream>>>(xb, src, dstp, e1w1t, ec1_b1,
                                                         e1w2t, ec1_b2, hbuf, E);
  k_f32_to_bf16<<<nbNH, THREADS, 0, stream>>>(hbuf, actb, N * 256);
  hipMemsetAsync(accb, 0, (size_t)N * 256 * 4, stream);
  k_edgeconv<256><<<(E + 63) / 64, THREADS, 0, stream>>>(actb, src, dstp, e2w1t, ec2_b1,
                                                         e2w2t, ec2_b2, accb, E);

  // ---- pool + head ----
  hipMemsetAsync(gpool, 0, (size_t)Gn * 512 * 4, stream);
  k_pool<<<(int)(((size_t)N * 512 + 255) / 256), THREADS, 0, stream>>>(xg, accb, batch, gpool, N);
  k_f32_to_bf16<<<(Gn * 512 + 255) / 256, THREADS, 0, stream>>>(gpool, gb, Gn * 512);
  const dim3 hGrid((Gn + 127) / 128, 4);
  k_gemm<<<hGrid, THREADS, 0, stream>>>(gb, fc1t, fc1_b, h1f, (__bf16*)nullptr, Gn, 512, 256, 1);
  k_head_out<<<1, 64, 0, stream>>>(h1f, out_w, out_b, outp, Gn);
}
